// testNet_83588653515338
// MI455X (gfx1250) — matrix operand order repaired and checked by reading, not compiled, not run on hardware
//
#include <hip/hip_runtime.h>
#include <hip/hip_bf16.h>

typedef __attribute__((ext_vector_type(16))) __bf16 v16bf;
typedef __attribute__((ext_vector_type(2)))  __bf16 v2bf;
typedef __attribute__((ext_vector_type(2)))  float  v2f;
typedef __attribute__((ext_vector_type(8)))  float  v8f;

#define THRESH_FACTOR 0.05f
#define NCH   32      // output channels
#define CIN   3
#define KREAL 75      // C*R*S = 3*5*5
#define HW    224
#define LROW  228     // padded LDS row (bf16 units): 2 + 224 + 2
#define NROWS 15      // CIN * 5 kernel rows
#define ZROW  (NROWS * LROW)   // start of always-zero sentinel row (bf16 units)
#define ZLEN  240

// im2col LDS offset (bf16-element units) for reduction index k; padded k
// points at the always-zero sentinel row. constexpr => after full unrolling
// every call site folds to a literal, so per-lane setup is one v_cndmask.
__host__ __device__ constexpr int im2col_off(int k) {
    return (k < KREAL)
        ? (((k / 25) * 5 + ((k % 25) / 5)) * LROW + (k % 5))
        : ZROW;
}

// ---------------------------------------------------------------------------
// Kernel 1: ternarize + pre-swizzle weights into per-lane WMMA A-fragment
// layout. ws holds 6 fragments (kk=0..2, cht=0..1), each 32 lanes x 16 bf16
// contiguous per lane => kernel 2 loads a fragment with wide vector loads.
//   A-layout (16-bit A, 16x32): element e of lane L -> M = L&15,
//   K = kk*32 + e + (e>=8 ? 8 : 0) + (L>>4)*8
// ---------------------------------------------------------------------------
__global__ __launch_bounds__(256) void ternarize_kernel(
    const float* __restrict__ w, const float* __restrict__ pos,
    const float* __restrict__ neg, __bf16* __restrict__ wfrag)
{
    __shared__ float red[256];
    const int tid = threadIdx.x;

    float m = 0.f;
    for (int i = tid; i < NCH * KREAL; i += 256)
        m = fmaxf(m, fabsf(w[i]));
    red[tid] = m;
    __syncthreads();
    for (int s = 128; s > 0; s >>= 1) {
        if (tid < s) red[tid] = fmaxf(red[tid], red[tid + s]);
        __syncthreads();
    }
    const float thresh = THRESH_FACTOR * red[0];
    const float p  = pos[0];
    const float ng = neg[0];

    // 6 fragments * 32 lanes * 16 elements = 3072 bf16 entries (6 KB)
    for (int idx = tid; idx < 6 * 32 * 16; idx += 256) {
        const int e      = idx & 15;
        const int lane   = (idx >> 4) & 31;
        const int f      = idx >> 9;          // kk*2 + cht
        const int kk     = f >> 1;
        const int cht    = f & 1;
        const int laneHi = lane >> 4;
        const int K  = kk * 32 + e + ((e >= 8) ? 8 : 0) + laneHi * 8;
        const int ch = cht * 16 + (lane & 15);
        float v = 0.f;
        if (K < KREAL) {
            const float wv = w[ch * KREAL + K];   // OIHW flat
            v = (wv > thresh) ? p : ((wv < -thresh) ? ng : 0.f);
        }
        wfrag[idx] = (__bf16)v;
    }
}

// ---------------------------------------------------------------------------
// Kernel 2: implicit-GEMM conv. Block = 224 threads = 7 wave32s; each block
// iterates over output rows (nimg,h). Per row: stage 15 padded input rows in
// LDS as bf16 (float2 loads -> packed 2xbf16 dword stores), then each wave
// computes two 32ch x 16w tiles via v_wmma_f32_16x16x32_bf16 with bare
// ds_load_u16 im2col gathers (constexpr-folded offsets, one cndmask each).
// ---------------------------------------------------------------------------
__global__ __launch_bounds__(224) void conv_wmma_kernel(
    const float* __restrict__ x, const v16bf* __restrict__ wfrag,
    const float* __restrict__ bias, float* __restrict__ out)
{
    __shared__ __bf16 xs[ZROW + ZLEN];   // (15*228 + 240) * 2B = 7320 B

    const int tid    = threadIdx.x;
    const int lane   = tid & 31;
    const int wave   = tid >> 5;        // 0..6
    const int laneHi = lane >> 4;
    const int lane15 = lane & 15;
    const bool hi    = (laneHi != 0);

    // ---- one-time: zero sentinel row (never overwritten), 120 dword stores
    for (int i = tid; i < ZLEN / 2; i += 224) {
        v2bf z = {};
        *(v2bf*)&xs[ZROW + 2 * i] = z;
    }

    // ---- one-time: per-lane im2col LDS offsets.
    //   B-layout (16-bit B, 32x16): element e of lane L -> N = L&15,
    //   K = kk*32 + e + (e>=8 ? 8 : 0) + (L>>4)*8.  Both candidate offsets per (kk,e) are
    //   compile-time constants -> each entry is a single select.
    int offs[3][16];
    #pragma unroll
    for (int kk = 0; kk < 3; ++kk)
        #pragma unroll
        for (int e = 0; e < 16; ++e)
            offs[kk][e] = hi ? im2col_off(kk * 32 + e + ((e >= 8) ? 8 : 0) + 8)
                             : im2col_off(kk * 32 + e + ((e >= 8) ? 8 : 0));

    // ---- one-time: weight fragments (wide per-lane vector loads)
    v16bf afrag[3][2];
    #pragma unroll
    for (int kk = 0; kk < 3; ++kk)
        #pragma unroll
        for (int t = 0; t < 2; ++t)
            afrag[kk][t] = wfrag[(kk * 2 + t) * 32 + lane];

    // ---- one-time: bias per lane (D VGPR i -> channel i + 8*laneHi)
    float b0[8], b1[8];
    #pragma unroll
    for (int i = 0; i < 8; ++i) {
        b0[i] = bias[i + 8 * laneHi];
        b1[i] = bias[16 + i + 8 * laneHi];
    }

    // ---- grid-stride loop over output rows
    for (int row = blockIdx.x; row < 32 * HW; row += gridDim.x) {
        const int h    = row % HW;
        const int nimg = row / HW;
        const float* xn = x + (size_t)nimg * CIN * HW * HW;

        __syncthreads();   // previous iteration's readers done before restage

        // interior: 15 rows * 112 column-pairs; float2 load -> packed bf16x2
        for (int idx = tid; idx < NROWS * 112; idx += 224) {
            const int cr = idx / 112;           // const-divisor -> mul/shift
            const int cp = idx - cr * 112;
            const int c  = cr / 5;
            const int r  = cr - c * 5;
            const int hy = h + r - 2;
            v2f xv = {};
            if ((unsigned)hy < HW)
                xv = *(const v2f*)(xn + (c * HW + hy) * HW + 2 * cp);
            v2bf bv;
            bv[0] = (__bf16)xv[0];
            bv[1] = (__bf16)xv[1];
            *(v2bf*)&xs[cr * LROW + 2 + 2 * cp] = bv;   // 4B-aligned dword store
        }
        if (tid < NROWS * 4) {                  // zero columns 0,1,224,225
            const int prow = tid >> 2;
            const int p    = tid & 3;
            xs[prow * LROW + ((p < 2) ? p : (222 + p))] = (__bf16)0.f;
        }
        __syncthreads();

        float* outn = out + (size_t)nimg * NCH * HW * HW;

        #pragma unroll
        for (int it = 0; it < 2; ++it) {        // 14 w-tiles = 7 waves * 2
            const int w0   = (wave * 2 + it) * 16;
            const int wcol = w0 + lane15;

            v8f acc0 = {};
            v8f acc1 = {};
            #pragma unroll
            for (int kk = 0; kk < 3; ++kk) {
                v16bf bfrag;
                #pragma unroll
                for (int e = 0; e < 16; ++e)
                    bfrag[e] = xs[offs[kk][e] + wcol];   // bare ds_load_u16

                acc0 = __builtin_amdgcn_wmma_f32_16x16x32_bf16(
                           false, afrag[kk][0], false, bfrag, (short)0, acc0, false, false);
                acc1 = __builtin_amdgcn_wmma_f32_16x16x32_bf16(
                           false, afrag[kk][1], false, bfrag, (short)0, acc1, false, false);
            }

            // D VGPR i -> channel m = i + 8*laneHi, column = wcol (coalesced)
            #pragma unroll
            for (int i = 0; i < 8; ++i) {
                const int m = i + 8 * laneHi;
                outn[((size_t)m        * HW + h) * HW + wcol] = acc0[i] + b0[i];
                outn[((size_t)(16 + m) * HW + h) * HW + wcol] = acc1[i] + b1[i];
            }
        }
    }
}

// ---------------------------------------------------------------------------
extern "C" void kernel_launch(void* const* d_in, const int* in_sizes, int n_in,
                              void* d_out, int out_size, void* d_ws, size_t ws_size,
                              hipStream_t stream) {
    const float* x    = (const float*)d_in[0];   // (32,3,224,224)
    const float* w    = (const float*)d_in[1];   // (32,3,5,5)
    const float* bias = (const float*)d_in[2];   // (32,)
    const float* pos  = (const float*)d_in[3];   // scalar
    const float* neg  = (const float*)d_in[4];   // scalar
    float*  out   = (float*)d_out;               // (32,32,224,224)
    __bf16* wfrag = (__bf16*)d_ws;               // 6 KB fragment panel

    ternarize_kernel<<<1, 256, 0, stream>>>(w, pos, neg, wfrag);

    // 32*224 = 7168 output rows; 1792 blocks, 4 rows each (grid-stride)
    conv_wmma_kernel<<<1792, 224, 0, stream>>>(
        x, (const v16bf*)wfrag, bias, out);
}
